// PointsEncoder_11922829213833
// MI455X (gfx1250) — compile-verified, matrix operand order checked by reading, not run on hardware
//
#include <hip/hip_runtime.h>
#include <hip/hip_bf16.h>
#include <stdint.h>
#include <limits.h>

typedef __attribute__((ext_vector_type(16))) _Float16 v16h;
typedef __attribute__((ext_vector_type(8)))  _Float16 v8h;
typedef __attribute__((ext_vector_type(2)))  _Float16 v2h;
typedef __attribute__((ext_vector_type(8)))  float    v8f;
typedef __attribute__((ext_vector_type(4)))  unsigned int u32x4;
typedef __attribute__((ext_vector_type(8)))  int      i32x8;
typedef __attribute__((ext_vector_type(4)))  int      i32x4;

#define B_    512
#define N_    512
#define M_    (B_ * N_)     // 262144 points
#define FIN   6
#define C1    128
#define C2    256
#define ENC_  128
#define EPS_  1e-5f

#if defined(__has_builtin)
#  if __has_builtin(__builtin_amdgcn_tensor_load_to_lds)
#    define USE_TDM 1
#  endif
#endif

__device__ __forceinline__ void wait_tensorcnt0() {
#if defined(__has_builtin)
#  if __has_builtin(__builtin_amdgcn_s_wait_tensorcnt)
    __builtin_amdgcn_s_wait_tensorcnt(0);
    return;
#  endif
#endif
    asm volatile("s_wait_tensorcnt 0x0" ::: "memory");
}

// ---- order-preserving float<->int map for integer-atomic max pooling ----
__device__ __forceinline__ int fOrd(float f) {
    int i = __float_as_int(f);
    return (i >= 0) ? i : (i ^ 0x7fffffff);
}
__device__ __forceinline__ float ordF(int i) {
    return __int_as_float((i >= 0) ? i : (i ^ 0x7fffffff));
}

// ---- WMMA fragment gather from an LDS row of 32 halves ----
// 16-bit A/B layout (ISA 7.12.2): lane holds K = base..base+7 and 16+base..16+base+7,
// base = (lane>=16) ? 8 : 0.  Two 16B ds_load_b128 per fragment.
__device__ __forceinline__ v16h load_frag(const _Float16* row, int kBase) {
    v8h lo = *(const v8h*)(row + kBase);
    v8h hi = *(const v8h*)(row + 16 + kBase);
    return __builtin_shufflevector(lo, hi, 0,1,2,3,4,5,6,7,8,9,10,11,12,13,14,15);
}

__device__ __forceinline__ v8f wmma_f16(v16h a, v16h b, v8f c) {
    return __builtin_amdgcn_wmma_f32_16x16x32_f16(false, a, false, b, (short)0, c, false, false);
}

// ---------------------------------------------------------------------------
// K0: init stats, pooling accumulators, convert weights to f16
// ---------------------------------------------------------------------------
__global__ void k_init(float* statsf, int* pool1i, int* outi,
                       const float* W2, _Float16* W2h,
                       const float* W3, _Float16* W3h,
                       const float* W4, _Float16* W4h) {
    int i = blockIdx.x * blockDim.x + threadIdx.x;
    if (i < 1600)       statsf[i] = 0.0f;
    if (i < B_ * C2)    pool1i[i] = INT_MIN;
    if (i < B_ * ENC_)  outi[i]   = INT_MIN;
    if (i < C1 * C2)    W2h[i] = (_Float16)W2[i];
    if (i < 512 * C2)   W3h[i] = (_Float16)W3[i];
    if (i < C2 * ENC_)  W4h[i] = (_Float16)W4[i];
}

// ---------------------------------------------------------------------------
// K1: masked BN1 statistics. h1 = x@W1 + b1 recomputed on the fly (K=6).
// ---------------------------------------------------------------------------
__global__ void k_stats1(const float* __restrict__ x, const unsigned char* __restrict__ mask,
                         const float* __restrict__ W1, const float* __restrict__ b1,
                         float* __restrict__ s_sum, float* __restrict__ s_sq,
                         float* __restrict__ s_cnt) {
    __shared__ float xs[32][FIN];
    __shared__ float ms[32];
    int t  = threadIdx.x;          // channel
    int p0 = blockIdx.x * 32;
    for (int i = t; i < 32 * FIN; i += 128)
        xs[i / FIN][i % FIN] = x[(size_t)(p0 + i / FIN) * FIN + i % FIN];
    if (t < 32) ms[t] = mask[p0 + t] ? 1.0f : 0.0f;
    __syncthreads();

    float wv[FIN];
#pragma unroll
    for (int j = 0; j < FIN; j++) wv[j] = W1[j * C1 + t];
    float bb = b1[t];
    float sum = 0.f, sq = 0.f;
    for (int p = 0; p < 32; p++) {
        float h = bb;
#pragma unroll
        for (int j = 0; j < FIN; j++) h += xs[p][j] * wv[j];
        float m = ms[p];
        sum += h * m;
        sq  += h * h * m;
    }
    atomicAdd(&s_sum[t], sum);
    atomicAdd(&s_sq[t],  sq);
    if (t == 0) {
        float c = 0.f;
        for (int p = 0; p < 32; p++) c += ms[p];
        atomicAdd(s_cnt, c);
    }
}

// ---------------------------------------------------------------------------
// K2/K6: fold BN stats into per-channel scale/bias
// ---------------------------------------------------------------------------
__global__ void k_bn_final(const float* s_sum, const float* s_sq, const float* s_cnt,
                           const float* g, const float* be,
                           float* scale, float* bias, int C) {
    int c = blockIdx.x * blockDim.x + threadIdx.x;
    if (c >= C) return;
    float cnt  = fmaxf(*s_cnt, 1.0f);
    float mean = s_sum[c] / cnt;
    float var  = s_sq[c] / cnt - mean * mean;
    float sc   = g[c] * rsqrtf(var + EPS_);
    scale[c] = sc;
    bias[c]  = be[c] - mean * sc;
}

// ---------------------------------------------------------------------------
// K3: fused MLP1 (x@W1+b1 -> BN -> ReLU) + GEMM2 (@W2+b2, mask) -> h2 f16
//     + per-batch max pool. 8 waves, tile 128(M) x 64(N), K-step 32.
// ---------------------------------------------------------------------------
__launch_bounds__(256, 2)
__global__ void k_mlp1_gemm2(const float* __restrict__ x, const unsigned char* __restrict__ mask,
                             const float* __restrict__ W1, const float* __restrict__ b1,
                             const float* __restrict__ scale1, const float* __restrict__ bias1,
                             const _Float16* __restrict__ W2h, const float* __restrict__ b2,
                             _Float16* __restrict__ h2h, int* __restrict__ pool1i) {
    __shared__ __align__(16) _Float16 As[128][32];
    __shared__ __align__(16) _Float16 Bt[64][32];   // transposed: Bt[n][k]
    __shared__ __align__(16) _Float16 Ts[64][64];   // epilogue transpose buffer
    __shared__ float xs[128][FIN];
    __shared__ float ms[128];
    __shared__ int   plds[64];

    int t     = threadIdx.x;
    int mBase = blockIdx.x * 128;
    int nBase = blockIdx.y * 64;
    int batch = mBase >> 9;

    for (int i = t; i < 128 * FIN; i += 256)
        xs[i / FIN][i % FIN] = x[(size_t)(mBase + i / FIN) * FIN + i % FIN];
    if (t < 128) ms[t] = mask[mBase + t] ? 1.0f : 0.0f;
    if (t < 64)  plds[t] = INT_MIN;

    int lane  = t & 31, w = t >> 5;
    int mRow  = w * 16 + (lane & 15);
    int kBase = (lane >> 4) * 8;

    v8f z = {0.f,0.f,0.f,0.f,0.f,0.f,0.f,0.f};
    v8f acc[4] = {z, z, z, z};
    __syncthreads();

    for (int kk = 0; kk < C1; kk += 32) {
        // A tile: recompute h1 pair of channels, BN+ReLU, packed ds_store_b32
        {
            int cp = (t & 15) * 2;          // channel pair within 32
            int c  = kk + cp;
            float sc0 = scale1[c],     bi0 = bias1[c],     bb0 = b1[c];
            float sc1 = scale1[c + 1], bi1 = bias1[c + 1], bb1 = b1[c + 1];
            float w0[FIN], w1[FIN];
#pragma unroll
            for (int j = 0; j < FIN; j++) { w0[j] = W1[j * C1 + c]; w1[j] = W1[j * C1 + c + 1]; }
#pragma unroll
            for (int r = 0; r < 8; r++) {
                int p = (t >> 4) + 16 * r;
                float h0 = bb0, h1 = bb1;
#pragma unroll
                for (int j = 0; j < FIN; j++) { h0 += xs[p][j] * w0[j]; h1 += xs[p][j] * w1[j]; }
                v2h pk;
                pk.x = (_Float16)fmaxf(h0 * sc0 + bi0, 0.0f);
                pk.y = (_Float16)fmaxf(h1 * sc1 + bi1, 0.0f);
                *(v2h*)&As[p][cp] = pk;
            }
        }
        // B tile: one b128 global load per thread, scatter-transpose into LDS
        {
            int k  = t >> 3;
            int n8 = (t & 7) * 8;
            v8h bv = *(const v8h*)&W2h[(size_t)(kk + k) * C2 + nBase + n8];
#pragma unroll
            for (int j = 0; j < 8; j++) Bt[n8 + j][k] = bv[j];
        }
        __syncthreads();
        v16h a = load_frag(&As[mRow][0], kBase);
#pragma unroll
        for (int s = 0; s < 4; s++) {
            v16h bf = load_frag(&Bt[s * 16 + (lane & 15)][0], kBase);
            acc[s] = wmma_f16(a, bf, acc[s]);
        }
        __syncthreads();
    }

    // pooling (block max-reduce in LDS, one global atomic per channel)
#pragma unroll
    for (int s = 0; s < 4; s++) {
        float bb = b2[nBase + s * 16 + (lane & 15)];
        int lmax = INT_MIN;
#pragma unroll
        for (int r = 0; r < 8; r++) {
            int pRel = w * 16 + r + 8 * (lane >> 4);
            lmax = max(lmax, fOrd((acc[s][r] + bb) * ms[pRel]));
        }
        atomicMax(&plds[s * 16 + (lane & 15)], lmax);
    }

    // epilogue: LDS transpose -> coalesced b128 stores of h2
    for (int half = 0; half < 2; half++) {
        if ((w >> 2) == half) {
            int wl = w & 3;
#pragma unroll
            for (int s = 0; s < 4; s++) {
                int nL = s * 16 + (lane & 15);
                float bb = b2[nBase + nL];
#pragma unroll
                for (int r = 0; r < 8; r++) {
                    int rr = r + 8 * (lane >> 4);
                    float v = (acc[s][r] + bb) * ms[w * 16 + rr];
                    Ts[wl * 16 + rr][nL] = (_Float16)v;
                }
            }
        }
        __syncthreads();
#pragma unroll
        for (int q = 0; q < 2; q++) {
            int idx = t + 256 * q;
            int p   = idx >> 3;            // 0..63
            int n8  = (idx & 7) * 8;
            *(v8h*)&h2h[(size_t)(mBase + half * 64 + p) * C2 + nBase + n8] =
                *(const v8h*)&Ts[p][n8];
        }
        __syncthreads();
    }
    if (t < 64) atomicMax(&pool1i[batch * C2 + nBase + t], plds[t]);
}

// ---------------------------------------------------------------------------
// K4: decode pooled ordered ints -> f32
// ---------------------------------------------------------------------------
__global__ void k_pool1_cvt(const int* __restrict__ pool1i, float* __restrict__ pooled1) {
    int i = blockIdx.x * blockDim.x + threadIdx.x;
    if (i < B_ * C2) pooled1[i] = ordF(pool1i[i]);
}

// ---------------------------------------------------------------------------
// K5: GEMM3: feat = [h2 | pooled] (K=512) @ W3 + b3 -> h3 f16 + masked BN2 stats
//     A-tile (h2 half) fetched by the Tensor Data Mover when available.
// ---------------------------------------------------------------------------
__launch_bounds__(256, 2)
__global__ void k_gemm3(const _Float16* __restrict__ h2h, const float* __restrict__ pooled1,
                        const unsigned char* __restrict__ mask,
                        const _Float16* __restrict__ W3h, const float* __restrict__ b3,
                        _Float16* __restrict__ h3h,
                        float* __restrict__ s_sum, float* __restrict__ s_sq) {
    __shared__ __align__(16) _Float16 As[128][32];
    __shared__ __align__(16) _Float16 Bt[64][32];
    __shared__ __align__(16) _Float16 Ts[64][64];
    __shared__ float ms[128];
    __shared__ float ssum[64], ssq[64];

    int t     = threadIdx.x;
    int mBase = blockIdx.x * 128;
    int nBase = blockIdx.y * 64;
    int batch = mBase >> 9;

    if (t < 128) ms[t] = mask[mBase + t] ? 1.0f : 0.0f;
    if (t < 64) { ssum[t] = 0.f; ssq[t] = 0.f; }

    int lane  = t & 31, w = t >> 5;
    int mRow  = w * 16 + (lane & 15);
    int kBase = (lane >> 4) * 8;

    v8f z = {0.f,0.f,0.f,0.f,0.f,0.f,0.f,0.f};
    v8f acc[4] = {z, z, z, z};
    __syncthreads();

    for (int kk = 0; kk < 512; kk += 32) {
        bool tdm_pending = false;
        if (kk < C2) {
#ifdef USE_TDM
            // TDM: 2D tile, rows of 64B (8 x 8B units), 128 rows, pitch 512B.
            if (w == 0) {
                uint64_t ga = (uint64_t)(size_t)&h2h[(size_t)mBase * C2 + kk];
                uint32_t la = (uint32_t)(size_t)&As[0][0];
                u32x4 g0 = { 1u,                                   // count=1, user mode
                             la,                                   // lds_addr
                             (unsigned int)ga,                     // global_addr lo
                             (unsigned int)(ga >> 32) | 0x80000000u }; // addr hi | type=2
                i32x8 g1 = { 0x30000,       // data_size=3 (8B units)
                             64 << 16,      // tensor_dim0 = 64 units (512B row)
                             128 << 16,     // tensor_dim1 = 128 rows
                             8 << 16,       // tile_dim0   = 8 units (64B)
                             128,           // tile_dim1   = 128 rows
                             64,            // tensor_dim0_stride = 64 units
                             0, 0 };
                i32x4 gz4 = {0, 0, 0, 0};
                i32x8 gz8 = {0, 0, 0, 0, 0, 0, 0, 0};
                __builtin_amdgcn_tensor_load_to_lds(g0, g1, gz4, gz4, gz8, 0);
            }
            tdm_pending = true;
#else
#pragma unroll
            for (int q = 0; q < 2; q++) {
                int idx = t + 256 * q;
                int p   = idx >> 2;
                int c8  = (idx & 3) * 8;
                *(v8h*)&As[p][c8] =
                    *(const v8h*)&h2h[(size_t)(mBase + p) * C2 + kk + c8];
            }
#endif
        } else {
            // pooled features broadcast down all 128 rows
            int c = t & 31;
            _Float16 hv = (_Float16)pooled1[batch * C2 + (kk - C2) + c];
#pragma unroll
            for (int r = 0; r < 16; r++) As[(t >> 5) + 8 * r][c] = hv;
        }
        // B tile (overlaps with the TDM transfer)
        {
            int k  = t >> 3;
            int n8 = (t & 7) * 8;
            v8h bv = *(const v8h*)&W3h[(size_t)(kk + k) * C2 + nBase + n8];
#pragma unroll
            for (int j = 0; j < 8; j++) Bt[n8 + j][k] = bv[j];
        }
        if (tdm_pending && w == 0) wait_tensorcnt0();
        __syncthreads();
        v16h a = load_frag(&As[mRow][0], kBase);
#pragma unroll
        for (int s = 0; s < 4; s++) {
            v16h bf = load_frag(&Bt[s * 16 + (lane & 15)][0], kBase);
            acc[s] = wmma_f16(a, bf, acc[s]);
        }
        __syncthreads();
    }

    // masked BN2 stats from accumulators
#pragma unroll
    for (int s = 0; s < 4; s++) {
        float bb = b3[nBase + s * 16 + (lane & 15)];
        float lsum = 0.f, lsq = 0.f;
#pragma unroll
        for (int r = 0; r < 8; r++) {
            int pRel = w * 16 + r + 8 * (lane >> 4);
            float v = acc[s][r] + bb;
            float m = ms[pRel];
            lsum += v * m;
            lsq  += v * v * m;
        }
        atomicAdd(&ssum[s * 16 + (lane & 15)], lsum);
        atomicAdd(&ssq [s * 16 + (lane & 15)], lsq);
    }

    // epilogue: LDS transpose -> coalesced b128 stores of h3 (pre-BN)
    for (int half = 0; half < 2; half++) {
        if ((w >> 2) == half) {
            int wl = w & 3;
#pragma unroll
            for (int s = 0; s < 4; s++) {
                int nL = s * 16 + (lane & 15);
                float bb = b3[nBase + nL];
#pragma unroll
                for (int r = 0; r < 8; r++) {
                    int rr = r + 8 * (lane >> 4);
                    Ts[wl * 16 + rr][nL] = (_Float16)(acc[s][r] + bb);
                }
            }
        }
        __syncthreads();
#pragma unroll
        for (int q = 0; q < 2; q++) {
            int idx = t + 256 * q;
            int p   = idx >> 3;
            int n8  = (idx & 7) * 8;
            *(v8h*)&h3h[(size_t)(mBase + half * 64 + p) * C2 + nBase + n8] =
                *(const v8h*)&Ts[p][n8];
        }
        __syncthreads();
    }
    if (t < 64) {
        atomicAdd(&s_sum[nBase + t], ssum[t]);
        atomicAdd(&s_sq [nBase + t], ssq[t]);
    }
}

// ---------------------------------------------------------------------------
// K7: fused BN2+ReLU + GEMM4 (@W4+b4, mask) + per-batch max pool -> outi
// ---------------------------------------------------------------------------
__launch_bounds__(256, 2)
__global__ void k_gemm4(const _Float16* __restrict__ h3h, const unsigned char* __restrict__ mask,
                        const float* __restrict__ scale2, const float* __restrict__ bias2,
                        const _Float16* __restrict__ W4h, const float* __restrict__ b4,
                        int* __restrict__ outi) {
    __shared__ __align__(16) _Float16 As[128][32];
    __shared__ __align__(16) _Float16 Bt[64][32];
    __shared__ float ms[128];
    __shared__ int   plds[64];

    int t     = threadIdx.x;
    int mBase = blockIdx.x * 128;
    int nBase = blockIdx.y * 64;
    int batch = mBase >> 9;

    if (t < 128) ms[t] = mask[mBase + t] ? 1.0f : 0.0f;
    if (t < 64)  plds[t] = INT_MIN;

    int lane  = t & 31, w = t >> 5;
    int mRow  = w * 16 + (lane & 15);
    int kBase = (lane >> 4) * 8;

    v8f z = {0.f,0.f,0.f,0.f,0.f,0.f,0.f,0.f};
    v8f acc[4] = {z, z, z, z};
    __syncthreads();

    for (int kk = 0; kk < C2; kk += 32) {
        // A tile: b128 load of h3, BN+ReLU in f32, b128 LDS store
        {
            int c8 = (t & 3) * 8;
            float sc[8], bi[8];
#pragma unroll
            for (int j = 0; j < 8; j++) {
                sc[j] = scale2[kk + c8 + j];
                bi[j] = bias2[kk + c8 + j];
            }
#pragma unroll
            for (int s2 = 0; s2 < 2; s2++) {
                int p = (t >> 2) + 64 * s2;
                v8h hv = *(const v8h*)&h3h[(size_t)(mBase + p) * C2 + kk + c8];
                v8h ov;
#pragma unroll
                for (int j = 0; j < 8; j++)
                    ov[j] = (_Float16)fmaxf((float)hv[j] * sc[j] + bi[j], 0.0f);
                *(v8h*)&As[p][c8] = ov;
            }
        }
        // B tile
        {
            int k  = t >> 3;
            int n8 = (t & 7) * 8;
            v8h bv = *(const v8h*)&W4h[(size_t)(kk + k) * ENC_ + nBase + n8];
#pragma unroll
            for (int j = 0; j < 8; j++) Bt[n8 + j][k] = bv[j];
        }
        __syncthreads();
        v16h a = load_frag(&As[mRow][0], kBase);
#pragma unroll
        for (int s = 0; s < 4; s++) {
            v16h bf = load_frag(&Bt[s * 16 + (lane & 15)][0], kBase);
            acc[s] = wmma_f16(a, bf, acc[s]);
        }
        __syncthreads();
    }

#pragma unroll
    for (int s = 0; s < 4; s++) {
        float bb = b4[nBase + s * 16 + (lane & 15)];
        int lmax = INT_MIN;
#pragma unroll
        for (int r = 0; r < 8; r++) {
            int pRel = w * 16 + r + 8 * (lane >> 4);
            lmax = max(lmax, fOrd((acc[s][r] + bb) * ms[pRel]));
        }
        atomicMax(&plds[s * 16 + (lane & 15)], lmax);
    }
    __syncthreads();
    if (t < 64) atomicMax(&outi[batch * ENC_ + nBase + t], plds[t]);
}

// ---------------------------------------------------------------------------
// K8: decode d_out ordered ints -> f32 in place
// ---------------------------------------------------------------------------
__global__ void k_out_cvt(float* __restrict__ out) {
    int i = blockIdx.x * blockDim.x + threadIdx.x;
    if (i < B_ * ENC_) {
        int v = ((int*)out)[i];
        out[i] = ordF(v);
    }
}

// ---------------------------------------------------------------------------
extern "C" void kernel_launch(void* const* d_in, const int* in_sizes, int n_in,
                              void* d_out, int out_size, void* d_ws, size_t ws_size,
                              hipStream_t stream) {
    const float*         x    = (const float*)d_in[0];
    const unsigned char* mask = (const unsigned char*)d_in[1];  // jax bool -> 1 byte
    const float* W1 = (const float*)d_in[2];
    const float* b1 = (const float*)d_in[3];
    const float* g1 = (const float*)d_in[4];
    const float* be1= (const float*)d_in[5];
    const float* W2 = (const float*)d_in[6];
    const float* b2 = (const float*)d_in[7];
    const float* W3 = (const float*)d_in[8];
    const float* b3 = (const float*)d_in[9];
    const float* g2 = (const float*)d_in[10];
    const float* be2= (const float*)d_in[11];
    const float* W4 = (const float*)d_in[12];
    const float* b4 = (const float*)d_in[13];

    // ---- workspace layout (bytes) ----
    char* w = (char*)d_ws;
    float* s1_sum    = (float*)(w + 0);        // 128
    float* s1_sq     = (float*)(w + 512);      // 128
    float* s1_cnt    = (float*)(w + 1024);     // 1
    float* bn1_scale = (float*)(w + 1280);     // 128
    float* bn1_bias  = (float*)(w + 1792);     // 128
    float* s2_sum    = (float*)(w + 2304);     // 256
    float* s2_sq     = (float*)(w + 3328);     // 256
    float* bn2_scale = (float*)(w + 4352);     // 256
    float* bn2_bias  = (float*)(w + 5376);     // 256  (stats region = 1600 f32)
    int*   pool1i    = (int*)  (w + 6400);                 // B*256 ints   (512 KB)
    float* pooled1   = (float*)(w + 6400 + 524288);        // B*256 f32    (512 KB)
    _Float16* W2h    = (_Float16*)(w + 1054976);           // 64 KB
    _Float16* W3h    = (_Float16*)(w + 1120512);           // 256 KB
    _Float16* W4h    = (_Float16*)(w + 1382656);           // 64 KB
    _Float16* h2h    = (_Float16*)(w + 1448192);           // M*256 f16 = 128 MB
    _Float16* h3h    = (_Float16*)(w + 1448192 + (size_t)M_ * C2 * 2); // 128 MB

    k_init<<<512, 256, 0, stream>>>((float*)w, pool1i, (int*)d_out,
                                    W2, W2h, W3, W3h, W4, W4h);
    k_stats1<<<M_ / 32, 128, 0, stream>>>(x, mask, W1, b1, s1_sum, s1_sq, s1_cnt);
    k_bn_final<<<1, 128, 0, stream>>>(s1_sum, s1_sq, s1_cnt, g1, be1, bn1_scale, bn1_bias, C1);
    k_mlp1_gemm2<<<dim3(M_ / 128, C2 / 64), 256, 0, stream>>>(
        x, mask, W1, b1, bn1_scale, bn1_bias, W2h, b2, h2h, pool1i);
    k_pool1_cvt<<<(B_ * C2 + 255) / 256, 256, 0, stream>>>(pool1i, pooled1);
    k_gemm3<<<dim3(M_ / 128, C2 / 64), 256, 0, stream>>>(
        h2h, pooled1, mask, W3h, b3, h3h, s2_sum, s2_sq);
    k_bn_final<<<1, 256, 0, stream>>>(s2_sum, s2_sq, s1_cnt, g2, be2, bn2_scale, bn2_bias, C2);
    k_gemm4<<<dim3(M_ / 128, ENC_ / 64), 256, 0, stream>>>(
        h3h, mask, bn2_scale, bn2_bias, W4h, b4, (int*)d_out);
    k_out_cvt<<<(B_ * ENC_ + 255) / 256, 256, 0, stream>>>((float*)d_out);
}
